// RegionProposalNetwork_34394098106610
// MI455X (gfx1250) — matrix operand order repaired and checked by reading, not compiled, not run on hardware
//
#include <hip/hip_runtime.h>
#include <hip/hip_bf16.h>
#include <math.h>

typedef unsigned int   u32t;
typedef unsigned short u16t;
typedef __attribute__((ext_vector_type(16))) __bf16 v16bf;
typedef __attribute__((ext_vector_type(8)))  float  v8f;
typedef __attribute__((ext_vector_type(4)))  unsigned int v4u;
typedef __attribute__((ext_vector_type(4)))  int    v4i;
typedef __attribute__((ext_vector_type(8)))  int    v8i;

#define HH      50
#define WWW     62
#define SPATIAL 3100          // 50*62
#define CIN     512
#define K1      4608          // 9*512, ordered k = (ky*3+kx)*512 + cin
#define NB      8
#define NANCH   27900         // 3100*9
#define NPOST   300
#define KSTEPS  144           // K1/32

// d_out offsets in float elements
#define OFF_LOCS   0
#define OFF_SCORES 892800     // NB*NANCH*4
#define OFF_ROIS   1339200    // + NB*NANCH*2
#define OFF_RIDX   1348800    // + NB*NPOST*4
#define OFF_ANCH   1351200    // + NB*NPOST

// workspace offsets (bytes)
#define WS_W1   0            // 512*4608*2        = 4,718,592
#define WS_W2   4718592      // 64*512*2          =    65,536
#define WS_XH   4784128      // 8*3100*512*2      = 25,395,200
#define WS_H    30179328     // 8*3100*512*2      = 25,395,200
#define WS_ROI  55574528     // 8*27900*4*4       =  3,571,200
#define WS_SC   59145728     // 8*27900*4         =    892,800

#if defined(__AMDGCN__) && __has_builtin(__builtin_amdgcn_tensor_load_to_lds) && \
    __has_builtin(__builtin_amdgcn_s_wait_tensorcnt)
#define USE_TDM 1
#else
#define USE_TDM 0
#endif

// ---------------- helpers ----------------

__device__ __forceinline__ u16t f2bf(float f) {
  u32t u = __builtin_bit_cast(u32t, f);
  u = u + 0x7FFFu + ((u >> 16) & 1u);   // round-to-nearest-even
  return (u16t)(u >> 16);
}

// LDS tile layout: row-major [rows][32 k], row stride 40 ushorts (80B, 16B aligned).
// A fragment (16x32 MxK, ISA 7.12.2): lane l -> m=l&15, g=l>>4; element e holds
// K = 16*(e>>3) + 8*g + (e&7)  => two contiguous 16B chunks at k=8g and k=16+8g.
__device__ __forceinline__ v16bf fragA(const u16t* base, int mi, int lane) {
  int m = lane & 15, g = lane >> 4;
  const uint4* p0 = (const uint4*)(base + (mi * 16 + m) * 40 + g * 8);
  const uint4* p1 = (const uint4*)(base + (mi * 16 + m) * 40 + 16 + g * 8);
  union { uint4 u[2]; v16bf v; } c; c.u[0] = *p0; c.u[1] = *p1; return c.v;
}
// B fragment (32x16 KxN): lane l -> n=l&15, g=l>>4; element e holds
// K = 16*(e>>3) + 8*g + (e&7) => two 16B chunks at k=8g and k=16+8g in an [n][k] tile.
__device__ __forceinline__ v16bf fragB(const u16t* base, int nj, int lane) {
  int nn = lane & 15, g = lane >> 4;
  const uint4* p0 = (const uint4*)(base + (nj * 16 + nn) * 40 + g * 8);
  const uint4* p1 = (const uint4*)(base + (nj * 16 + nn) * 40 + 16 + g * 8);
  union { uint4 u[2]; v16bf v; } c; c.u[0] = *p0; c.u[1] = *p1; return c.v;
}

// im2col B gather: one 8-cin contiguous chunk, fixed (ky,kx), zero at borders
__device__ __forceinline__ uint4 gatherB(const u16t* xb, int kg, int y, int x, bool sval) {
  int r = kg >> 9, cin0 = kg & 511;
  int ky = r / 3, kx = r - ky * 3;
  int iy = y + ky - 1, ix = x + kx - 1;
  uint4 b = {0u, 0u, 0u, 0u};
  if (sval && iy >= 0 && iy < HH && ix >= 0 && ix < WWW)
    b = *(const uint4*)(xb + ((size_t)iy * WWW + ix) * CIN + cin0);
  return b;
}

#if USE_TDM
// TDM: 2D tile load, 128 rows x 32 bf16 (64B), row stride 4608 elements,
// LDS padding: 16B after every 64B => 80B LDS row stride (matches frag layout).
__device__ __forceinline__ void tdm_load_A(const u16t* gsrc, const u16t* lds_dst) {
  unsigned long long ga = (unsigned long long)(size_t)gsrc;
  v4u g0 = { 1u,                                   // count=1 (valid user descriptor)
             (u32t)(size_t)lds_dst,                // lds_addr (low 32 bits of generic)
             (u32t)ga,                             // global_addr[31:0]
             (u32t)((ga >> 32) & 0x01FFFFFFu) | 0x80000000u };  // addr[56:32] | type=2
  v8i g1 = { (int)0x06D10000,                      // data_size=2B, pad_en, int=16dw, amt=4dw
             (int)0xFFFF0000u,                     // tensor_dim0 lo (huge, no OOB)
             (int)0xFFFF7FFFu,                     // tensor_dim0 hi | tensor_dim1 lo
             (int)0x00207FFF,                      // tensor_dim1 hi | tile_dim0=32
             128,                                  // tile_dim1=128, tile_dim2=0
             4608,                                 // tensor_dim0_stride lo
             0, 0 };
  v4i g2 = { 0x7FFFFFFF, 0, 0, 0 };
  v4i g3 = { 0, 0, 0, 0 };
#if __clang_major__ >= 23
  v8i g4 = { 0, 0, 0, 0, 0, 0, 0, 0 };
  __builtin_amdgcn_tensor_load_to_lds(g0, g1, g2, g3, g4, 0);
#else
  __builtin_amdgcn_tensor_load_to_lds(g0, g1, g2, g3, 0);
#endif
}
#endif

// ---------------- prep kernels ----------------

// conv1_w OIHW fp32 -> bf16 [cout][(ky*3+kx)*512 + cin]
__global__ void k_w1perm(const float* __restrict__ w, u16t* __restrict__ wb) {
  int i = blockIdx.x * 256 + threadIdx.x;   // 512*4608 exact
  int cout = i / K1;
  int rem  = i - cout * K1;
  int r    = rem >> 9;        // 0..8 = ky*3+kx
  int cin  = rem & 511;
  wb[i] = f2bf(w[((size_t)cout * CIN + cin) * 9 + r]);
}

// x NCHW fp32 -> bf16 NHWC [n][s][cin]
__global__ void k_xh(const float* __restrict__ x, u16t* __restrict__ xh) {
  int i = blockIdx.x * 256 + threadIdx.x;   // 8*3100*512 exact (49600 blocks)
  int cin = i & 511;
  int ns  = i >> 9;
  int s   = ns % SPATIAL;
  int n   = ns / SPATIAL;
  xh[i] = f2bf(x[((size_t)n * CIN + cin) * SPATIAL + s]);
}

__global__ void k_w2build(const float* __restrict__ loc_w,
                          const float* __restrict__ score_w,
                          u16t* __restrict__ w2) {
  int i = blockIdx.x * 256 + threadIdx.x;   // 64*512 exact
  int row = i >> 9, k = i & 511;
  float v = 0.0f;
  if (row < 36)      v = loc_w[row * 512 + k];
  else if (row < 54) v = score_w[(row - 36) * 512 + k];
  w2[i] = f2bf(v);
}

// ---------------- conv1 3x3 as implicit GEMM (bf16 WMMA, 128x128 tiles) ----------------
// grid (25 Ntiles, 4 Mtiles, 8 batch), 256 threads = 8 waves.
// Wave w: mi = {2(w&3), 2(w&3)+1}, nj = (w>>2)*4 .. +3  => 8 WMMAs / K-step.
// A tiles arrive via TDM (double-buffered); B tiles via guarded b128 im2col gather.

__global__ __launch_bounds__(256)
void k_conv3x3_wmma(const u16t* __restrict__ xh, const u16t* __restrict__ w1,
                    const float* __restrict__ b1, u16t* __restrict__ hout) {
  __shared__ __align__(16) u16t As[2][128 * 40];
  __shared__ __align__(16) u16t Bs[128 * 40];
  const int n     = blockIdx.z;
  const int cout0 = blockIdx.y * 128;
  const int s0    = blockIdx.x * 128;
  const int t     = threadIdx.x;
  const int wave  = t >> 5, lane = t & 31;
  const int mi0   = (wave & 3) * 2;
  const int nj0   = (wave >> 2) * 4;
  const int row0  = t >> 2;            // 0..63
  const int koff  = (t & 3) * 8;       // 0,8,16,24

  const int sB0 = s0 + row0;
  const int sB1 = sB0 + 64;
  const int yB0 = sB0 / WWW, xB0 = sB0 - yB0 * WWW;
  const int yB1 = sB1 / WWW, xB1 = sB1 - yB1 * WWW;
  const bool v0 = sB0 < SPATIAL, v1 = sB1 < SPATIAL;

  const u16t* xb = xh + (size_t)n * SPATIAL * CIN;
  const u16t* wA = w1 + (size_t)cout0 * K1;

  v8f acc[2][4] = {};

  // prologue: B regs for tile 0; A tile 0 in flight / staged
  uint4 breg0 = gatherB(xb, koff, yB0, xB0, v0);
  uint4 breg1 = gatherB(xb, koff, yB1, xB1, v1);
#if USE_TDM
  if (wave == 0) tdm_load_A(wA, &As[0][0]);
#else
  {
    uint4 a0 = *(const uint4*)(wA + (size_t)row0 * K1 + koff);
    uint4 a1 = *(const uint4*)(wA + (size_t)(row0 + 64) * K1 + koff);
    *(uint4*)(&As[0][0] + row0 * 40 + koff)        = a0;
    *(uint4*)(&As[0][0] + (row0 + 64) * 40 + koff) = a1;
  }
  uint4 areg0 = *(const uint4*)(wA + (size_t)row0 * K1 + 32 + koff);
  uint4 areg1 = *(const uint4*)(wA + (size_t)(row0 + 64) * K1 + 32 + koff);
#endif

  for (int it = 0; it < KSTEPS; ++it) {
    const int buf = it & 1;
    const u16t* Asb = &As[buf][0];
    __syncthreads();                     // X: everyone done reading As[buf^1], Bs
    *(uint4*)(Bs + row0 * 40 + koff)        = breg0;
    *(uint4*)(Bs + (row0 + 64) * 40 + koff) = breg1;
#if USE_TDM
    if (wave == 0) {
      if (it + 1 < KSTEPS) {
        tdm_load_A(wA + (size_t)(it + 1) * 32, &As[buf ^ 1][0]);
        __builtin_amdgcn_s_wait_tensorcnt(1);   // tile `it` resident
      } else {
        __builtin_amdgcn_s_wait_tensorcnt(0);
      }
    }
#else
    if (it + 1 < KSTEPS) {
      *(uint4*)(&As[buf ^ 1][0] + row0 * 40 + koff)        = areg0;
      *(uint4*)(&As[buf ^ 1][0] + (row0 + 64) * 40 + koff) = areg1;
    }
#endif
    __syncthreads();                     // Y: As[buf] + Bs ready for all waves
    if (it + 1 < KSTEPS) {
      int kg = (it + 1) * 32 + koff;
      breg0 = gatherB(xb, kg, yB0, xB0, v0);
      breg1 = gatherB(xb, kg, yB1, xB1, v1);
#if !USE_TDM
      if (it + 2 < KSTEPS) {
        areg0 = *(const uint4*)(wA + (size_t)row0 * K1 + (it + 2) * 32 + koff);
        areg1 = *(const uint4*)(wA + (size_t)(row0 + 64) * K1 + (it + 2) * 32 + koff);
      }
#endif
    }
    v16bf af0 = fragA(Asb, mi0, lane);
    v16bf af1 = fragA(Asb, mi0 + 1, lane);
#pragma unroll
    for (int j = 0; j < 4; ++j) {
      v16bf bfj = fragB(Bs, nj0 + j, lane);
      acc[0][j] = __builtin_amdgcn_wmma_f32_16x16x32_bf16(false, af0, false, bfj,
                                                          (short)0, acc[0][j], false, false);
      acc[1][j] = __builtin_amdgcn_wmma_f32_16x16x32_bf16(false, af1, false, bfj,
                                                          (short)0, acc[1][j], false, false);
    }
  }

  // epilogue: bias + ReLU, pack 8 consecutive couts -> one b128 store
  const int g = lane >> 4, nn = lane & 15;
#pragma unroll
  for (int i2 = 0; i2 < 2; ++i2) {
#pragma unroll
    for (int j = 0; j < 4; ++j) {
      int ss = s0 + (nj0 + j) * 16 + nn;
      if (ss >= SPATIAL) continue;
      int coutb = cout0 + (mi0 + i2) * 16 + 8 * g;
      union { uint4 u; u16t h[8]; } pk;
#pragma unroll
      for (int v = 0; v < 8; ++v) {
        float va = acc[i2][j][v] + b1[coutb + v];
        pk.h[v] = f2bf(va > 0.0f ? va : 0.0f);
      }
      *(uint4*)(hout + ((size_t)n * SPATIAL + ss) * CIN + coutb) = pk.u;
    }
  }
}

// ---------------- head 1x1 convs as GEMM (64x3100xK512) ----------------
// grid (49, 1, 8), 256 threads. Rows 0..35 = loc channels, 36..53 = score channels.

__global__ __launch_bounds__(256)
void k_head_wmma(const u16t* __restrict__ h, const u16t* __restrict__ w2,
                 const float* __restrict__ loc_b, const float* __restrict__ sc_b,
                 float* __restrict__ out) {
  __shared__ __align__(16) u16t As[64 * 40];
  __shared__ __align__(16) u16t Bs[64 * 40];
  const int n    = blockIdx.z;
  const int s0   = blockIdx.x * 64;
  const int t    = threadIdx.x;
  const int wave = t >> 5, lane = t & 31;
  const int mi   = wave & 3;
  const int nj0  = (wave >> 2) * 2;
  const int m_ld = t >> 2;
  const int k_ld = (t & 3) * 8;
  const int s    = s0 + m_ld;

  v8f acc0 = {}; v8f acc1 = {};

  for (int kt = 0; kt < 512; kt += 32) {
    uint4 av = *(const uint4*)(w2 + (size_t)m_ld * 512 + kt + k_ld);
    uint4 bvv = {0u, 0u, 0u, 0u};
    if (s < SPATIAL)
      bvv = *(const uint4*)(h + ((size_t)n * SPATIAL + s) * 512 + kt + k_ld);
    __syncthreads();
    *(uint4*)(As + m_ld * 40 + k_ld) = av;
    *(uint4*)(Bs + m_ld * 40 + k_ld) = bvv;
    __syncthreads();

    v16bf af  = fragA(As, mi, lane);
    v16bf bf0 = fragB(Bs, nj0, lane);
    v16bf bf1 = fragB(Bs, nj0 + 1, lane);
    acc0 = __builtin_amdgcn_wmma_f32_16x16x32_bf16(false, af, false, bf0,
                                                   (short)0, acc0, false, false);
    acc1 = __builtin_amdgcn_wmma_f32_16x16x32_bf16(false, af, false, bf1,
                                                   (short)0, acc1, false, false);
  }

  const int g = lane >> 4, nn = lane & 15;
#pragma unroll
  for (int v = 0; v < 8; ++v) {
    int c2 = mi * 16 + v + 8 * g;
#pragma unroll
    for (int j = 0; j < 2; ++j) {
      int ss = s0 + (nj0 + j) * 16 + nn;
      if (ss >= SPATIAL) continue;
      float val = (j == 0) ? acc0[v] : acc1[v];
      if (c2 < 36) {                     // rpn_locs: channel a*4 + q at (y,x)
        val += loc_b[c2];
        int a = c2 >> 2, q = c2 & 3;
        out[OFF_LOCS + ((size_t)n * NANCH + (size_t)ss * 9 + a) * 4 + q] = val;
      } else if (c2 < 54) {              // rpn_scores: channel a*2 + q
        int ch = c2 - 36;
        val += sc_b[ch];
        int a = ch >> 1, q = ch & 1;
        out[OFF_SCORES + ((size_t)n * NANCH + (size_t)ss * 9 + a) * 2 + q] = val;
      }
    }
  }
}

// ---------------- anchors ----------------

__global__ void k_anchor(float* __restrict__ out) {
  int idx = blockIdx.x * 256 + threadIdx.x;
  if (idx >= NANCH) return;
  int sp = idx / 9, a = idx - sp * 9;
  int y = sp / WWW, x = sp - y * WWW;
  int ri = a / 3, si = a - ri * 3;
  float ratio = (ri == 0) ? 0.5f : ((ri == 1) ? 1.0f : 2.0f);
  float sc    = (si == 0) ? 8.0f : ((si == 1) ? 16.0f : 32.0f);
  float hh = 16.0f * sc * sqrtf(ratio);
  float ww = 16.0f * sc * sqrtf(1.0f / ratio);
  float oy = y * 16.0f + 8.0f, ox = x * 16.0f + 8.0f;
  float* o = out + OFF_ANCH + (size_t)idx * 4;
  o[0] = oy - 0.5f * hh; o[1] = ox - 0.5f * ww;
  o[2] = oy + 0.5f * hh; o[3] = ox + 0.5f * ww;
}

// ---------------- fg softmax + loc2bbox + clip + min-size ----------------

__global__ void k_roi(const float* __restrict__ out_ro,
                      const int* __restrict__ img_h, const int* __restrict__ img_w,
                      float* __restrict__ roi_ws, float* __restrict__ sc_ws) {
  int idx = blockIdx.x * 256 + threadIdx.x;
  if (idx >= NB * NANCH) return;
  int i = idx % NANCH;
  const float* loc = out_ro + OFF_LOCS + (size_t)idx * 4;
  const float* scp = out_ro + OFF_SCORES + (size_t)idx * 2;
  const float* an  = out_ro + OFF_ANCH + (size_t)i * 4;
  float fg = 1.0f / (1.0f + expf(scp[0] - scp[1]));   // softmax[...,1]
  float a0 = an[0], a1 = an[1], a2 = an[2], a3 = an[3];
  float ha = a2 - a0, wa = a3 - a1;
  float cy = a0 + 0.5f * ha, cx = a1 + 0.5f * wa;
  float ncy = loc[0] * ha + cy, ncx = loc[1] * wa + cx;
  float nh = expf(loc[2]) * ha, nw = expf(loc[3]) * wa;
  float Hc = (float)img_h[0], Wc = (float)img_w[0];
  float r0 = fminf(fmaxf(ncy - 0.5f * nh, 0.0f), Hc);
  float r1 = fminf(fmaxf(ncx - 0.5f * nw, 0.0f), Wc);
  float r2 = fminf(fmaxf(ncy + 0.5f * nh, 0.0f), Hc);
  float r3 = fminf(fmaxf(ncx + 0.5f * nw, 0.0f), Wc);
  bool keep = ((r2 - r0) >= 16.0f) && ((r3 - r1) >= 16.0f);
  float* rw = roi_ws + (size_t)idx * 4;
  rw[0] = r0; rw[1] = r1; rw[2] = r2; rw[3] = r3;
  sc_ws[idx] = keep ? fg : -__builtin_inff();
}

// ---------------- iterative argmax NMS (one block per batch) ----------------

__global__ __launch_bounds__(1024)
void k_nms(const float* __restrict__ roi_ws, float* __restrict__ sc_ws,
           float* __restrict__ out) {
  __shared__ float smax[1024];
  __shared__ int   sidx[1024];
  __shared__ float sbox[4];
  __shared__ int   sok;
  const int n = blockIdx.x, t = threadIdx.x;
  float* sc = sc_ws + (size_t)n * NANCH;
  const float* boxes = roi_ws + (size_t)n * NANCH * 4;

  for (int it = 0; it < NPOST; ++it) {
    float best = -__builtin_inff(); int bi = 0;
    for (int i = t; i < NANCH; i += 1024) {
      float v = sc[i];
      if (v > best) { best = v; bi = i; }
    }
    smax[t] = best; sidx[t] = bi;
    __syncthreads();
    for (int off = 512; off > 0; off >>= 1) {
      if (t < off && smax[t + off] > smax[t]) {
        smax[t] = smax[t + off]; sidx[t] = sidx[t + off];
      }
      __syncthreads();
    }
    if (t == 0) {
      int j  = sidx[0];
      int ok = (smax[0] > -__builtin_inff()) ? 1 : 0;
      float b0 = boxes[j * 4 + 0], b1 = boxes[j * 4 + 1];
      float b2 = boxes[j * 4 + 2], b3 = boxes[j * 4 + 3];
      sbox[0] = b0; sbox[1] = b1; sbox[2] = b2; sbox[3] = b3;
      sok = ok;
      float* o = out + OFF_ROIS + ((size_t)n * NPOST + it) * 4;
      o[0] = ok ? b0 : 0.0f; o[1] = ok ? b1 : 0.0f;
      o[2] = ok ? b2 : 0.0f; o[3] = ok ? b3 : 0.0f;
      sc[j] = -__builtin_inff();
    }
    __syncthreads();
    if (sok) {
      float b0 = sbox[0], b1 = sbox[1], b2 = sbox[2], b3 = sbox[3];
      float area1 = (b2 - b0) * (b3 - b1);
      for (int i = t; i < NANCH; i += 1024) {
        float c0 = boxes[i * 4 + 0], c1 = boxes[i * 4 + 1];
        float c2 = boxes[i * 4 + 2], c3 = boxes[i * 4 + 3];
        float ty = fmaxf(b0, c0), tx = fmaxf(b1, c1);
        float by = fminf(b2, c2), bx = fminf(b3, c3);
        float inter = fmaxf(by - ty, 0.0f) * fmaxf(bx - tx, 0.0f);
        float area2 = (c2 - c0) * (c3 - c1);
        float iou = inter / (area1 + area2 - inter + 1e-10f);
        if (iou > 0.7f) sc[i] = -__builtin_inff();
      }
    }
    __syncthreads();
  }
}

__global__ void k_ridx(void* __restrict__ out) {
  int i = blockIdx.x * 256 + threadIdx.x;
  if (i < NB * NPOST) ((int*)out)[OFF_RIDX + i] = i / NPOST;
}

// ---------------- launch ----------------

extern "C" void kernel_launch(void* const* d_in, const int* in_sizes, int n_in,
                              void* d_out, int out_size, void* d_ws, size_t ws_size,
                              hipStream_t stream) {
  const float* x       = (const float*)d_in[0];
  const float* conv1_w = (const float*)d_in[1];
  const float* conv1_b = (const float*)d_in[2];
  const float* score_w = (const float*)d_in[3];
  const float* score_b = (const float*)d_in[4];
  const float* loc_w   = (const float*)d_in[5];
  const float* loc_b   = (const float*)d_in[6];
  const int*   img_h   = (const int*)d_in[7];
  const int*   img_w   = (const int*)d_in[8];
  float* out = (float*)d_out;

  char* ws = (char*)d_ws;
  u16t*  w1_bf  = (u16t*)(ws + WS_W1);
  u16t*  w2_bf  = (u16t*)(ws + WS_W2);
  u16t*  xh_bf  = (u16t*)(ws + WS_XH);
  u16t*  h_bf   = (u16t*)(ws + WS_H);
  float* roi_ws = (float*)(ws + WS_ROI);
  float* sc_ws  = (float*)(ws + WS_SC);

  k_w1perm<<<9216, 256, 0, stream>>>(conv1_w, w1_bf);       // 512*4608/256
  k_xh<<<49600, 256, 0, stream>>>(x, xh_bf);                // 8*3100*512/256
  k_w2build<<<128, 256, 0, stream>>>(loc_w, score_w, w2_bf);
  k_conv3x3_wmma<<<dim3(25, 4, NB), 256, 0, stream>>>(xh_bf, w1_bf, conv1_b, h_bf);
  k_head_wmma<<<dim3(49, 1, NB), 256, 0, stream>>>(h_bf, w2_bf, loc_b, score_b, out);
  k_anchor<<<(NANCH + 255) / 256, 256, 0, stream>>>(out);
  k_roi<<<(NB * NANCH + 255) / 256, 256, 0, stream>>>(out, img_h, img_w, roi_ws, sc_ws);
  k_nms<<<NB, 1024, 0, stream>>>(roi_ws, sc_ws, out);
  k_ridx<<<(NB * NPOST + 255) / 256, 256, 0, stream>>>(d_out);
}
